// MultiHeadAttention_7627861917746
// MI455X (gfx1250) — matrix operand order repaired and checked by reading, not compiled, not run on hardware
//
#include <hip/hip_runtime.h>
#include <hip/hip_bf16.h>

typedef __attribute__((ext_vector_type(16))) _Float16 v16h;
typedef __attribute__((ext_vector_type(8)))  _Float16 v8h;
typedef __attribute__((ext_vector_type(8)))  float    v8f;

constexpr int BS  = 4;
constexpr int SEQ = 1024;
constexpr int DM  = 1024;
constexpr int NH  = 16;
constexpr int DK  = 64;
constexpr int TOK = BS * SEQ;           // 4096 tokens
constexpr int LDT = 40;                 // padded LDS row stride (halves), mult of 8

// ---------- WMMA helpers (wave32, 16x16x32 f16 -> f32) ----------

__device__ __forceinline__ v8f wmma_f16(v16h a, v16h b, v8f c) {
  return __builtin_amdgcn_wmma_f32_16x16x32_f16(false, a, false, b, (short)0, c,
                                                false, false);
}

// A fragment: 16x32 (MxK), source is row-major MxK with K contiguous.
// lanes 0-15: M=lane, K {0..7} then {16..23}; lanes 16-31: M=lane-16, K {8..15},{24..31}
__device__ __forceinline__ v16h frag_a(const _Float16* mat, int row, int k, int ld) {
  const int lane = threadIdx.x & 31;
  const int m    = lane & 15;
  const int kb   = (lane >> 4) << 3;          // 0 or 8
  const _Float16* p = mat + (size_t)(row + m) * ld + k + kb;
  v8h lo = *(const v8h*)p;
  v8h hi = *(const v8h*)(p + 16);
  return __builtin_shufflevector(lo, hi, 0,1,2,3,4,5,6,7,8,9,10,11,12,13,14,15);
}

// B fragment: 32x16 (KxN), source stored N-major: row n holds K contiguous.
// lanes 0-15: N=lane, K {0..7} then {16..23}; lanes 16-31: N=lane-16, K {8..15},{24..31}
__device__ __forceinline__ v16h frag_b(const _Float16* mat, int col, int k, int ld) {
  const int lane = threadIdx.x & 31;
  const int n    = lane & 15;
  const int kb   = (lane >> 4) << 3;          // 0 or 8
  const _Float16* p = mat + (size_t)(col + n) * ld + k + kb;
  v8h lo = *(const v8h*)p;
  v8h hi = *(const v8h*)(p + 16);
  return __builtin_shufflevector(lo, hi, 0,1,2,3,4,5,6,7,8,9,10,11,12,13,14,15);
}

__device__ __forceinline__ v8h cvt8(float4 f0, float4 f1) {
  v8h h;
  h[0] = (_Float16)f0.x; h[1] = (_Float16)f0.y; h[2] = (_Float16)f0.z; h[3] = (_Float16)f0.w;
  h[4] = (_Float16)f1.x; h[5] = (_Float16)f1.y; h[6] = (_Float16)f1.z; h[7] = (_Float16)f1.w;
  return h;
}

// ---------- Kernel 1: QKV projection  C = X @ W^T + b  (f32 in, f16 out) ----------
// grid: (TOK/128, DM/128, 3), block 256 (8 waves, 2x4 wave grid, 64x32 per wave)
__global__ __launch_bounds__(256) void qkv_proj_kernel(
    const float* __restrict__ q, const float* __restrict__ k, const float* __restrict__ v,
    const float* __restrict__ Wq, const float* __restrict__ Wk, const float* __restrict__ Wv,
    const float* __restrict__ bq, const float* __restrict__ bk, const float* __restrict__ bv,
    _Float16* __restrict__ qp, _Float16* __restrict__ kp, _Float16* __restrict__ vpT) {
  const int which = blockIdx.z;
  const float* X    = (which == 0) ? q  : (which == 1) ? k  : v;
  const float* W    = (which == 0) ? Wq : (which == 1) ? Wk : Wv;
  const float* bias = (which == 0) ? bq : (which == 1) ? bk : bv;

  __shared__ __align__(16) _Float16 sA[128 * LDT];
  __shared__ __align__(16) _Float16 sB[128 * LDT];

  const int tid  = threadIdx.x;
  const int lane = tid & 31;
  const int wave = tid >> 5;
  const int wm   = wave >> 2;      // 0..1  (64 rows each)
  const int wn   = wave & 3;       // 0..3  (32 cols each)
  const int m0   = blockIdx.x * 128;
  const int n0   = blockIdx.y * 128;
  const int fln  = lane & 15;
  const int moff = (lane >> 4) << 3;

  v8f acc[4][2];
  #pragma unroll
  for (int fn = 0; fn < 2; ++fn) {
    float bval = bias[n0 + wn * 32 + fn * 16 + fln];
    #pragma unroll
    for (int fm = 0; fm < 4; ++fm)
      #pragma unroll
      for (int r = 0; r < 8; ++r) acc[fm][fn][r] = bval;
  }

  for (int kk = 0; kk < DM; kk += 32) {
    {  // stage A tile 128x32 (f32 -> f16)
      int row = tid >> 1;
      int cb  = (tid & 1) * 16;
      const float4* gp = (const float4*)(X + (size_t)(m0 + row) * DM + kk + cb);
      *(v8h*)&sA[row * LDT + cb]     = cvt8(gp[0], gp[1]);
      *(v8h*)&sA[row * LDT + cb + 8] = cvt8(gp[2], gp[3]);
    }
    {  // stage B tile 128x32 from W (already N-major x K)
      int row = tid >> 1;
      int cb  = (tid & 1) * 16;
      const float4* gp = (const float4*)(W + (size_t)(n0 + row) * DM + kk + cb);
      *(v8h*)&sB[row * LDT + cb]     = cvt8(gp[0], gp[1]);
      *(v8h*)&sB[row * LDT + cb + 8] = cvt8(gp[2], gp[3]);
    }
    __syncthreads();

    v16h bf0 = frag_b(sB, wn * 32,      0, LDT);
    v16h bf1 = frag_b(sB, wn * 32 + 16, 0, LDT);
    #pragma unroll
    for (int fm = 0; fm < 4; ++fm) {
      v16h af = frag_a(sA, wm * 64 + fm * 16, 0, LDT);
      acc[fm][0] = wmma_f16(af, bf0, acc[fm][0]);
      acc[fm][1] = wmma_f16(af, bf1, acc[fm][1]);
    }
    __syncthreads();
  }

  #pragma unroll
  for (int fm = 0; fm < 4; ++fm)
    #pragma unroll
    for (int fn = 0; fn < 2; ++fn) {
      int gm = m0 + wm * 64 + fm * 16 + moff;
      int gn = n0 + wn * 32 + fn * 16 + fln;
      if (which < 2) {
        _Float16* O = (which == 0) ? qp : kp;
        #pragma unroll
        for (int r = 0; r < 8; ++r)
          O[(size_t)(gm + r) * DM + gn] = (_Float16)acc[fm][fn][r];
      } else {
        int h = gn >> 6, d = gn & 63;  // vpT[b][h][d][s]
        #pragma unroll
        for (int r = 0; r < 8; ++r) {
          int M = gm + r, b = M >> 10, s = M & 1023;
          vpT[(((size_t)(b * NH + h) * DK + d) << 10) + s] = (_Float16)acc[fm][fn][r];
        }
      }
    }
}

// ---------- Kernel 2: fused scores + softmax ----------
// P = softmax((Qh @ Kh^T)/8) written directly; S never touches HBM.
// grid: (SEQ/64, 1, BS*NH), block 256. Block owns a 64-row x 1024-col strip;
// wave w owns cols [w*128, w*128+128): 4x8 = 32 accum fragments per wave.
__global__ __launch_bounds__(256) void scores_softmax_kernel(
    const _Float16* __restrict__ qp, const _Float16* __restrict__ kp,
    float* __restrict__ att) {
  const int z = blockIdx.z;
  const int b = z >> 4, h = z & 15;
  const _Float16* A = qp + (size_t)b * SEQ * DM + h * DK;
  const _Float16* B = kp + (size_t)b * SEQ * DM + h * DK;
  float* C = att + (size_t)z * SEQ * SEQ;

  const int tid  = threadIdx.x;
  const int lane = tid & 31;
  const int wave = tid >> 5;
  const int m0   = blockIdx.x * 64;
  const int ncol = wave * 128;         // this wave's column base
  const int fln  = lane & 15;
  const int moff = (lane >> 4) << 3;   // 0 or 8

  __shared__ float redbuf[64][8];      // per-row, per-wave partials
  __shared__ float rowfin[64];         // row max, later 1/rowsum

  v8f acc[4][8];
  #pragma unroll
  for (int fm = 0; fm < 4; ++fm)
    #pragma unroll
    for (int fn = 0; fn < 8; ++fn)
      #pragma unroll
      for (int r = 0; r < 8; ++r) acc[fm][fn][r] = 0.0f;

  // GEMM: K = 64 in two k-steps, fragments straight from global (L2-resident)
  #pragma unroll
  for (int ks = 0; ks < DK; ks += 32) {
    v16h af[4];
    #pragma unroll
    for (int fm = 0; fm < 4; ++fm) af[fm] = frag_a(A, m0 + fm * 16, ks, DM);
    #pragma unroll
    for (int fn = 0; fn < 8; ++fn) {
      v16h bf = frag_b(B, ncol + fn * 16, ks, DM);
      #pragma unroll
      for (int fm = 0; fm < 4; ++fm) acc[fm][fn] = wmma_f16(af[fm], bf, acc[fm][fn]);
    }
  }

  // scale by 1/sqrt(dk) = 1/8
  #pragma unroll
  for (int fm = 0; fm < 4; ++fm)
    #pragma unroll
    for (int fn = 0; fn < 8; ++fn)
      #pragma unroll
      for (int r = 0; r < 8; ++r) acc[fm][fn][r] *= 0.125f;

  // ---- row max: lane-local over fn, shfl over the 16-lane row group, LDS over waves
  #pragma unroll
  for (int fm = 0; fm < 4; ++fm)
    #pragma unroll
    for (int r = 0; r < 8; ++r) {
      float mval = acc[fm][0][r];
      #pragma unroll
      for (int fn = 1; fn < 8; ++fn) mval = fmaxf(mval, acc[fm][fn][r]);
      #pragma unroll
      for (int off = 8; off >= 1; off >>= 1)
        mval = fmaxf(mval, __shfl_xor(mval, off, 32));
      if (fln == 0) redbuf[fm * 16 + moff + r][wave] = mval;
    }
  __syncthreads();
  if (tid < 64) {
    float m = redbuf[tid][0];
    #pragma unroll
    for (int w = 1; w < 8; ++w) m = fmaxf(m, redbuf[tid][w]);
    rowfin[tid] = m;
  }
  __syncthreads();

  // ---- exp + row sum
  #pragma unroll
  for (int fm = 0; fm < 4; ++fm)
    #pragma unroll
    for (int r = 0; r < 8; ++r) {
      const int row = fm * 16 + moff + r;
      const float rm = rowfin[row];
      float ls = 0.0f;
      #pragma unroll
      for (int fn = 0; fn < 8; ++fn) {
        float e = __expf(acc[fm][fn][r] - rm);
        acc[fm][fn][r] = e;
        ls += e;
      }
      #pragma unroll
      for (int off = 8; off >= 1; off >>= 1) ls += __shfl_xor(ls, off, 32);
      if (fln == 0) redbuf[row][wave] = ls;
    }
  __syncthreads();
  if (tid < 64) {
    float s = redbuf[tid][0];
    #pragma unroll
    for (int w = 1; w < 8; ++w) s += redbuf[tid][w];
    rowfin[tid] = 1.0f / s;
  }
  __syncthreads();

  // ---- normalize and store probabilities
  #pragma unroll
  for (int fm = 0; fm < 4; ++fm)
    #pragma unroll
    for (int r = 0; r < 8; ++r) {
      const int row = fm * 16 + moff + r;
      const float inv = rowfin[row];
      #pragma unroll
      for (int fn = 0; fn < 8; ++fn)
        C[(size_t)(m0 + row) * SEQ + ncol + fn * 16 + fln] = acc[fm][fn][r] * inv;
    }
}

// ---------- Kernel 3: O = P @ Vh  (P f32 staged->f16, Vh^T already f16) ----------
// grid: (SEQ/128, 1, BS*NH), block 256
__global__ __launch_bounds__(256) void av_kernel(
    const float* __restrict__ att, const _Float16* __restrict__ vpT,
    float* __restrict__ attnout) {
  const int z = blockIdx.z;
  const int b = z >> 4, h = z & 15;
  const float*    P  = att + (size_t)z * SEQ * SEQ;
  const _Float16* Bv = vpT + (size_t)z * DK * SEQ;   // [d][s], s contiguous

  __shared__ __align__(16) _Float16 sA[128 * LDT];

  const int tid  = threadIdx.x;
  const int lane = tid & 31;
  const int wave = tid >> 5;
  const int wm   = wave >> 1, wn = wave & 1;
  const int m0   = blockIdx.x * 128;
  const int fln  = lane & 15;
  const int moff = (lane >> 4) << 3;

  v8f acc[2][2];
  #pragma unroll
  for (int i = 0; i < 2; ++i)
    #pragma unroll
    for (int j = 0; j < 2; ++j)
      #pragma unroll
      for (int r = 0; r < 8; ++r) acc[i][j][r] = 0.0f;

  for (int kk = 0; kk < SEQ; kk += 32) {
    {  // stage P tile 128x32 f32 -> f16
      int row = tid >> 1;
      int cb  = (tid & 1) * 16;
      const float4* gp = (const float4*)(P + (size_t)(m0 + row) * SEQ + kk + cb);
      *(v8h*)&sA[row * LDT + cb]     = cvt8(gp[0], gp[1]);
      *(v8h*)&sA[row * LDT + cb + 8] = cvt8(gp[2], gp[3]);
    }
    __syncthreads();
    v16h af0 = frag_a(sA, wm * 32,      0, LDT);
    v16h af1 = frag_a(sA, wm * 32 + 16, 0, LDT);
    v16h bf0 = frag_b(Bv, wn * 32,      kk, SEQ);
    v16h bf1 = frag_b(Bv, wn * 32 + 16, kk, SEQ);
    acc[0][0] = wmma_f16(af0, bf0, acc[0][0]);
    acc[0][1] = wmma_f16(af0, bf1, acc[0][1]);
    acc[1][0] = wmma_f16(af1, bf0, acc[1][0]);
    acc[1][1] = wmma_f16(af1, bf1, acc[1][1]);
    __syncthreads();
  }

  #pragma unroll
  for (int fm = 0; fm < 2; ++fm)
    #pragma unroll
    for (int fn = 0; fn < 2; ++fn) {
      int gm = m0 + wm * 32 + fm * 16 + moff;          // q position in batch b
      int gn = wn * 32 + fn * 16 + fln;                // 0..63 within head
      #pragma unroll
      for (int r = 0; r < 8; ++r)
        attnout[(size_t)(b * SEQ + gm + r) * DM + h * DK + gn] = acc[fm][fn][r];
    }
}

// ---------- Kernel 4: residual + LayerNorm ----------
__global__ __launch_bounds__(256) void ln_kernel(
    const float* __restrict__ q_in, const float* __restrict__ attnout,
    const float* __restrict__ gamma, const float* __restrict__ beta,
    float* __restrict__ out) {
  const size_t row = blockIdx.x;
  const int t = threadIdx.x;
  __shared__ float red[256];

  float4 a = ((const float4*)(q_in    + row * DM))[t];
  float4 o = ((const float4*)(attnout + row * DM))[t];
  float4 x; x.x = a.x + o.x; x.y = a.y + o.y; x.z = a.z + o.z; x.w = a.w + o.w;

  red[t] = x.x + x.y + x.z + x.w;
  __syncthreads();
  for (int s = 128; s > 0; s >>= 1) {
    if (t < s) red[t] += red[t + s];
    __syncthreads();
  }
  const float mean = red[0] * (1.0f / DM);
  __syncthreads();
  red[t] = x.x * x.x + x.y * x.y + x.z * x.z + x.w * x.w;
  __syncthreads();
  for (int s = 128; s > 0; s >>= 1) {
    if (t < s) red[t] += red[t + s];
    __syncthreads();
  }
  const float var  = red[0] * (1.0f / DM) - mean * mean;
  const float rstd = rsqrtf(var + 1e-5f);

  float4 g  = ((const float4*)gamma)[t];
  float4 bt = ((const float4*)beta)[t];
  float4 y;
  y.x = (x.x - mean) * rstd * g.x + bt.x;
  y.y = (x.y - mean) * rstd * g.y + bt.y;
  y.z = (x.z - mean) * rstd * g.z + bt.z;
  y.w = (x.w - mean) * rstd * g.w + bt.w;
  ((float4*)(out + row * DM))[t] = y;
}

// ---------- Host launch ----------
extern "C" void kernel_launch(void* const* d_in, const int* in_sizes, int n_in,
                              void* d_out, int out_size, void* d_ws, size_t ws_size,
                              hipStream_t stream) {
  const float* q     = (const float*)d_in[0];
  const float* k     = (const float*)d_in[1];
  const float* v     = (const float*)d_in[2];
  const float* Wq    = (const float*)d_in[3];
  const float* bq    = (const float*)d_in[4];
  const float* Wk    = (const float*)d_in[5];
  const float* bk    = (const float*)d_in[6];
  const float* Wv    = (const float*)d_in[7];
  const float* bv    = (const float*)d_in[8];
  const float* gamma = (const float*)d_in[9];
  const float* beta  = (const float*)d_in[10];
  // head (d_in[11]) is fixed at 16 per the reference setup.

  // workspace: qp (f16), kp (f16), vpT (f16), attnout (f32)  ~42 MB
  _Float16* qp   = (_Float16*)d_ws;
  _Float16* kp   = qp  + (size_t)TOK * DM;
  _Float16* vpT  = kp  + (size_t)TOK * DM;
  float* attnout = (float*)(vpT + (size_t)TOK * DM);

  // outputs: [normed (TOK*DM)] ++ [att_score (BS*NH*SEQ*SEQ)]
  float* out_norm = (float*)d_out;
  float* att      = out_norm + (size_t)TOK * DM;

  dim3 blk(256);

  qkv_proj_kernel<<<dim3(TOK / 128, DM / 128, 3), blk, 0, stream>>>(
      q, k, v, Wq, Wk, Wv, bq, bk, bv, qp, kp, vpT);

  scores_softmax_kernel<<<dim3(SEQ / 64, 1, BS * NH), blk, 0, stream>>>(qp, kp, att);

  av_kernel<<<dim3(SEQ / 128, 1, BS * NH), blk, 0, stream>>>(att, vpT, attnout);

  ln_kernel<<<dim3(TOK), blk, 0, stream>>>(q, attnout, gamma, beta, out_norm);
}
